// Baseline_28561532518446
// MI455X (gfx1250) — matrix operand order repaired and checked by reading, not compiled, not run on hardware
//
#include <hip/hip_runtime.h>

typedef __attribute__((ext_vector_type(16))) _Float16 v16h;
typedef __attribute__((ext_vector_type(8)))  float    v8f;

#define VR      8
#define BINS    512      // VR^3
#define NPTS    8192
#define BATCH   1024
#define CLASSES 40
#define NWAVES  8        // 256 threads / wave32

// ---------------------------------------------------------------------------
// Stage 1: per-sample voxel histogram. 1 block = 1 sample. Per-wave privatized
// LDS counters (8 x 512 u32 = 16 KB << 320 KB/WGP) to reduce ds_add_u32
// contention; streamed coalesced reads of x (100 MB total -> HBM-bound,
// ~4.3 us at 23.3 TB/s). Output: normalized counts as f16 (exact for k*2^-13),
// which is directly the WMMA A-matrix for stage 2.
// ---------------------------------------------------------------------------
__global__ __launch_bounds__(256) void hist_kernel(const float* __restrict__ x,
                                                   _Float16* __restrict__ hcounts) {
  __shared__ unsigned int cnt[NWAVES][BINS];
  const int b    = blockIdx.x;
  const int tid  = threadIdx.x;
  const int wave = tid >> 5;

  unsigned int* flat = &cnt[0][0];
  for (int i = tid; i < NWAVES * BINS; i += 256) flat[i] = 0u;
  __syncthreads();

  const float* xp = x + (size_t)b * NPTS * 3;

#pragma unroll 4
  for (int i = 0; i < NPTS / 256; ++i) {
    const int p = i * 256 + tid;
    const float x0 = xp[p * 3 + 0];
    const float x1 = xp[p * 3 + 1];
    const float x2 = xp[p * 3 + 2];
    const bool valid = (x0 >= -2.0f) && (x0 <= 2.0f) &&
                       (x1 >= -2.0f) && (x1 <= 2.0f) &&
                       (x2 >= -2.0f) && (x2 <= 2.0f);
    if (valid) {
      int i0 = (int)floorf((x0 + 2.0f) * 2.0f); i0 = i0 > (VR - 1) ? (VR - 1) : i0;
      int i1 = (int)floorf((x1 + 2.0f) * 2.0f); i1 = i1 > (VR - 1) ? (VR - 1) : i1;
      int i2 = (int)floorf((x2 + 2.0f) * 2.0f); i2 = i2 > (VR - 1) ? (VR - 1) : i2;
      const int lin = (i0 * VR + i1) * VR + i2;
      atomicAdd(&cnt[wave][lin], 1u);          // ds_add_u32
    }
  }
  __syncthreads();

  const float inv = 1.0f / (float)NPTS;
  for (int i = tid; i < BINS; i += 256) {
    unsigned int s = 0u;
#pragma unroll
    for (int w = 0; w < NWAVES; ++w) s += cnt[w][i];
    hcounts[(size_t)b * BINS + i] = (_Float16)((float)s * inv);
  }
}

// ---------------------------------------------------------------------------
// Stage 2: out(1024x40) = counts(1024x512) @ W^T(512x40) + b via
// v_wmma_f32_16x16x32_f16. One wave per 16x16 output tile; 64 M-tiles x
// 3 N-tiles (40 padded to 48; W row index clamped so EXEC stays all-ones
// through the WMMA loop per ISA restriction). K=512 -> 16 WMMAs per wave.
//
// Fragment layouts (ISA 7.12.2, wave32):
//  A 16x32 f16 : lane = M (lane&15); halfs j<8 -> K = (lane>>4)*8 + j,
//                j>=8 -> K = 16 + (lane>>4)*8 + (j-8).
//  B 32x16 f16 : lane = N (lane&15); half j -> same K as A's half j.
//  C/D 16x16 f32: reg r -> M = r + 8*(lane>>4), N = lane&15.
// ---------------------------------------------------------------------------
__global__ __launch_bounds__(32) void gemm_kernel(const _Float16* __restrict__ A,
                                                  const float* __restrict__ W,
                                                  const float* __restrict__ bias,
                                                  float* __restrict__ out) {
  const int tile = blockIdx.x;          // 0..191
  const int mt   = tile / 3;
  const int nt   = tile - mt * 3;
  const int lane = threadIdx.x;         // 0..31
  const int h    = lane >> 4;           // lane half
  const int sub  = lane & 15;

  const int arow = mt * 16 + sub;                         // < 1024 always
  const int bcol = nt * 16 + sub;                         // 0..47
  const int bc   = bcol < CLASSES ? bcol : CLASSES - 1;   // clamp, keep EXEC full

  const _Float16* ap = A + (size_t)arow * BINS;
  const float*    wp = W + (size_t)bc   * BINS;

  v8f acc = {};
#pragma unroll
  for (int k0 = 0; k0 < BINS; k0 += 32) {
    v16h a, bb;
#pragma unroll
    for (int j = 0; j < 8; ++j) {
      a[j]     = ap[k0 + h * 8 + j];
      a[8 + j] = ap[k0 + 16 + h * 8 + j];
    }
#pragma unroll
    for (int j = 0; j < 8; ++j) {
      bb[j]     = (_Float16)wp[k0 + h * 8 + j];
      bb[8 + j] = (_Float16)wp[k0 + 16 + h * 8 + j];
    }
    acc = __builtin_amdgcn_wmma_f32_16x16x32_f16(
        /*neg_a=*/false, a, /*neg_b=*/false, bb,
        /*c_mod=*/(short)0, acc, /*reuse_a=*/false, /*reuse_b=*/false);
  }

  if (bcol < CLASSES) {
    const float bv = bias[bcol];
#pragma unroll
    for (int r = 0; r < 8; ++r) {
      const int crow = mt * 16 + r + 8 * h;
      out[(size_t)crow * CLASSES + bcol] = acc[r] + bv;
    }
  }
}

// ---------------------------------------------------------------------------
extern "C" void kernel_launch(void* const* d_in, const int* in_sizes, int n_in,
                              void* d_out, int out_size, void* d_ws, size_t ws_size,
                              hipStream_t stream) {
  const float* x  = (const float*)d_in[0];   // (1024, 8192, 3) f32
  const float* W  = (const float*)d_in[1];   // (40, 512) f32
  const float* bv = (const float*)d_in[2];   // (40,) f32
  float* out      = (float*)d_out;           // (1024, 40) f32

  _Float16* hbuf = (_Float16*)d_ws;          // 1024*512 f16 = 1 MB scratch

  hist_kernel<<<BATCH, 256, 0, stream>>>(x, hbuf);
  gemm_kernel<<<(BATCH / 16) * 3, 32, 0, stream>>>(hbuf, W, bv, out);
}
